// DenseGAT_17489106829442
// MI455X (gfx1250) — matrix operand order repaired and checked by reading, not compiled, not run on hardware
//
#include <hip/hip_runtime.h>
#include <hip/hip_bf16.h>
#include <math.h>

// Dense GAT, fused flash-style. MI455X / gfx1250, wave32, WMMA f16->f32,
// async global->LDS double-buffered staging.
#define BB   8
#define LL   2048
#define CIN  256
#define COUT 256
#define NH   4
#define HD   64
#define NEG_SLOPE 0.2f
#define FILLV (-3.402823466e+38f)   // jnp.finfo(float32).min
#define NJT  (LL / 64)              // 32 j-tiles of 64

typedef __attribute__((ext_vector_type(16))) _Float16 v16h;
typedef __attribute__((ext_vector_type(8)))  _Float16 v8h;
typedef __attribute__((ext_vector_type(8)))  float    v8f;

// ---- async global->LDS support (guarded; host pass sees 0) ----------------
#if defined(__AMDGCN__) && __has_builtin(__builtin_amdgcn_global_load_async_to_lds_b128)
#define HAS_ASYNC 1
#else
#define HAS_ASYNC 0
#endif

#if HAS_ASYNC
// builtin prototype (from hipcc diagnostic): pointee is int __vector_size__(16),
// param0 in AS1 (printed "__device__"), param1 in AS3.
typedef int v4i_gl __attribute__((__vector_size__(16)));
#define GPTR(p) ((__attribute__((address_space(1))) v4i_gl*)(__attribute__((address_space(1))) void*)(void*)(p))
#define LPTR(p) ((__attribute__((address_space(3))) v4i_gl*)(__attribute__((address_space(3))) void*)(void*)(p))
#if __has_builtin(__builtin_amdgcn_s_wait_asynccnt)
#define WAIT_ASYNC(n) __builtin_amdgcn_s_wait_asynccnt(n)
#else
#define WAIT_ASYNC(n) asm volatile("s_wait_asynccnt " #n ::: "memory")
#endif
#else
#define WAIT_ASYNC(n) ((void)0)
#endif

// ---------------------------------------------------------------------------
// Kernel 1: h = x @ W^T  (M=16384, N=256, K=256); one 16x16 D tile per wave,
// 8 chained v_wmma_f32_16x16x32_f16.  Stored transposed f16: hT[b][head][d][l].
// ---------------------------------------------------------------------------
__global__ __launch_bounds__(256) void gat_proj(const float* __restrict__ x,
                                                const float* __restrict__ W,
                                                _Float16* __restrict__ hT) {
  const int wid   = blockIdx.x * 8 + (threadIdx.x >> 5);
  const int lane  = threadIdx.x & 31;
  const int row0  = (wid >> 4) * 16;   // global M index = b*L + l
  const int col0  = (wid & 15) * 16;
  const int mN    = lane & 15;
  const int g     = lane >> 4;

  const float* __restrict__ arow = x + (size_t)(row0 + mN) * CIN;
  const float* __restrict__ brow = W + (size_t)(col0 + mN) * CIN;

  v8f c = {};
  for (int k0 = 0; k0 < CIN; k0 += 32) {
    v16h a, b;
#pragma unroll
    for (int v = 0; v < 8; ++v) {
      const int ka = k0 + ((v & 4) << 2) + (g << 3) + ((v & 3) << 1);
      a[2 * v]     = (_Float16)arow[ka];
      a[2 * v + 1] = (_Float16)arow[ka + 1];
      const int kb = k0 + ((v & 4) << 2) + (g << 3) + ((v & 3) << 1);
      b[2 * v]     = (_Float16)brow[kb];
      b[2 * v + 1] = (_Float16)brow[kb + 1];
    }
    c = __builtin_amdgcn_wmma_f32_16x16x32_f16(false, a, false, b,
                                               (short)0, c, false, false);
  }
#pragma unroll
  for (int r = 0; r < 8; ++r) {
    const int grow = row0 + r + (g << 3);      // b*L + l (8 consecutive l/lane)
    const int bidx = grow >> 11;
    const int l    = grow & (LL - 1);
    const int o    = col0 + mN;
    hT[(((size_t)bidx * NH + (o >> 6)) * HD + (o & 63)) * LL + l] = (_Float16)c[r];
  }
}

// ---------------------------------------------------------------------------
// Kernel 2: a_src/a_dst per-node attention scalars (coalesced in l).
// ---------------------------------------------------------------------------
__global__ __launch_bounds__(256) void gat_attvec(const _Float16* __restrict__ hT,
                                                  const float* __restrict__ att_src,
                                                  const float* __restrict__ att_dst,
                                                  float* __restrict__ a_src,
                                                  float* __restrict__ a_dst) {
  const int idx  = blockIdx.x * 256 + threadIdx.x;   // over B*H*L
  const int l    = idx & (LL - 1);
  const int bh   = idx >> 11;
  const int head = bh & (NH - 1);
  const _Float16* __restrict__ base = hT + (size_t)bh * HD * LL + l;
  float s = 0.f, t = 0.f;
#pragma unroll 4
  for (int d = 0; d < HD; ++d) {
    const float v = (float)base[(size_t)d * LL];
    s += v * att_src[head * HD + d];
    t += v * att_dst[head * HD + d];
  }
  a_src[idx] = s;
  a_dst[idx] = t;
}

// ---------------------------------------------------------------------------
// Kernel 3: fused masked-softmax attention + out = alpha @ h.
// Block (256 thr = 8 waves) per (b, head, 32-row i-tile); j-step 64.
// Pass 1: two-sweep (max, then sum-of-exp), __shfl_xor reductions in-wave.
// Pass 2: async double-buffered h staging, P tile f16 in LDS, 2 chained
//         WMMAs per wave per barrier pair, b128 fragment loads.
// ---------------------------------------------------------------------------
__global__ __launch_bounds__(256) void gat_attn(const unsigned char* __restrict__ adj,
                                                const _Float16* __restrict__ hT,
                                                const float* __restrict__ a_src,
                                                const float* __restrict__ a_dst,
                                                float* __restrict__ out) {
  __shared__ _Float16 sP [32][72];      // P tile: [i-row][j (64)], padded
  __shared__ _Float16 sHT[2][64][72];   // double-buffered hT tile: [d][j (64)]

  const int blk   = blockIdx.x;         // B*H*(L/32) = 2048
  const int itile = blk & 63;
  const int bh    = blk >> 6;
  const int head  = bh & (NH - 1);
  const int b     = bh >> 2;
  const int i0    = itile * 32;
  const int tid   = threadIdx.x;

  const float* __restrict__ ad = a_dst + (size_t)bh * LL;
  const _Float16* __restrict__ hbase = hT + (size_t)bh * HD * LL;

  // stage one 64x64 h tile (transposed layout) into buffer bb.
  // thread t: row d = t>>2, 2 x b128 chunks at jc = (t&3)*16 (+8).
  auto stage = [&](int j0, int bb) {
    const int d  = tid >> 2;
    const int jc = (tid & 3) << 4;
    const _Float16* src = hbase + (size_t)d * LL + j0 + jc;
    _Float16*       dst = &sHT[bb][d][jc];
#if HAS_ASYNC
    __builtin_amdgcn_global_load_async_to_lds_b128(GPTR(src), LPTR(dst), 0, 0);
    __builtin_amdgcn_global_load_async_to_lds_b128(GPTR(src), LPTR(dst), 16, 0);
#else
    *(v8h*)dst       = *(const v8h*)src;
    *(v8h*)(dst + 8) = *(const v8h*)(src + 8);
#endif
  };

  stage(0, 0);   // prologue: overlap tile-0 DMA with pass 1

  // ---------- pass 1: row max, then sum of exp (shuffle-reduced) ----------
  const int row = tid >> 3;             // 0..31 (8-thread groups, wave-local)
  const int sub = tid & 7;
  const float asr = a_src[(size_t)bh * LL + i0 + row];
  const unsigned char* __restrict__ mrow = adj + ((size_t)b * LL + i0 + row) * LL;

  float mx = -__builtin_inff();
  for (int jc = sub * 4; jc < LL; jc += 32) {
    const unsigned int mk = *(const unsigned int*)(mrow + jc);
    const float4 dv = *(const float4*)(ad + jc);
    const float d4[4] = {dv.x, dv.y, dv.z, dv.w};
#pragma unroll
    for (int q = 0; q < 4; ++q) {
      float e = asr + d4[q];
      e = (e >= 0.f) ? e : NEG_SLOPE * e;
      e = ((mk >> (8 * q)) & 0xffu) ? e : FILLV;
      mx = fmaxf(mx, e);
    }
  }
#pragma unroll
  for (int o = 1; o < 8; o <<= 1) mx = fmaxf(mx, __shfl_xor(mx, o));

  float S = 0.f;
  for (int jc = sub * 4; jc < LL; jc += 32) {
    const unsigned int mk = *(const unsigned int*)(mrow + jc);
    const float4 dv = *(const float4*)(ad + jc);
    const float d4[4] = {dv.x, dv.y, dv.z, dv.w};
#pragma unroll
    for (int q = 0; q < 4; ++q) {
      float e = asr + d4[q];
      e = (e >= 0.f) ? e : NEG_SLOPE * e;
      e = ((mk >> (8 * q)) & 0xffu) ? e : FILLV;
      S += __expf(e - mx);
    }
  }
#pragma unroll
  for (int o = 1; o < 8; o <<= 1) S += __shfl_xor(S, o);

  const float Mr = mx;          // row max (finite; >= FILLV)
  const float Ir = 1.0f / S;    // S >= 1 always

  // ---------- pass 2: WMMA accumulation over j (step 64, pipelined) -------
  const int wave  = tid >> 5;
  const int lane  = tid & 31;
  const int rtile = wave >> 2;     // rows [0,16) / [16,32)
  const int ntile = wave & 3;      // d-slice
  const int mN    = lane & 15;
  const int g     = lane >> 4;

  v8f acc = {};
  const v8h* __restrict__ arowP = (const v8h*)&sP[(rtile << 4) + mN][0];

  for (int jt = 0; jt < NJT; ++jt) {
    const int j0 = jt * 64;
    if (jt + 1 < NJT) stage(j0 + 64, (jt + 1) & 1);   // prefetch next tile

    // compute P tile (overlaps with async DMA): row = tid>>3, 8 consecutive j
    {
      const int jb = (tid & 7) << 3;
      const unsigned int mk0 = *(const unsigned int*)(mrow + j0 + jb);
      const unsigned int mk1 = *(const unsigned int*)(mrow + j0 + jb + 4);
      const float4 dv0 = *(const float4*)(ad + j0 + jb);
      const float4 dv1 = *(const float4*)(ad + j0 + jb + 4);
      const float d8[8] = {dv0.x, dv0.y, dv0.z, dv0.w, dv1.x, dv1.y, dv1.z, dv1.w};
#pragma unroll
      for (int q = 0; q < 8; ++q) {
        float e = asr + d8[q];
        e = (e >= 0.f) ? e : NEG_SLOPE * e;
        const unsigned int mb = (q < 4) ? ((mk0 >> (8 * q)) & 0xffu)
                                        : ((mk1 >> (8 * (q - 4))) & 0xffu);
        e = mb ? e : FILLV;
        sP[row][jb + q] = (_Float16)(__expf(e - Mr) * Ir);
      }
    }

    if (jt + 1 < NJT) { WAIT_ASYNC(2); } else { WAIT_ASYNC(0); }
    __syncthreads();

    // two chained WMMAs (j halves) with b128 fragment loads from LDS
    const v8h* __restrict__ browH = (const v8h*)&sHT[jt & 1][(ntile << 4) + mN][0];
#pragma unroll
    for (int hb = 0; hb < 2; ++hb) {
      const v8h alo = arowP[4 * hb + g];          // A: K = 32hb + 8g .. +7
      const v8h ahi = arowP[4 * hb + g + 2];      // A: K = 32hb + 16 + 8g .. +7
      const v8h blo = browH[4 * hb + g];          // B: K = 32hb + 8g .. +7
      const v8h bhi = browH[4 * hb + g + 2];      // B: K = 32hb + 16 + 8g .. +7
      v16h a, bf;
#pragma unroll
      for (int e = 0; e < 8; ++e) {
        a [e] = alo[e];  a [8 + e] = ahi[e];
        bf[e] = blo[e];  bf[8 + e] = bhi[e];
      }
      acc = __builtin_amdgcn_wmma_f32_16x16x32_f16(false, a, false, bf,
                                                   (short)0, acc, false, false);
    }
    __syncthreads();
  }

  // store: out[b][l][head*HD + d]
#pragma unroll
  for (int r = 0; r < 8; ++r) {
    const int li = i0 + (rtile << 4) + r + (g << 3);
    const int o  = head * HD + (ntile << 4) + mN;
    out[((size_t)b * LL + li) * COUT + o] = acc[r];
  }
}

// ---------------------------------------------------------------------------
extern "C" void kernel_launch(void* const* d_in, const int* in_sizes, int n_in,
                              void* d_out, int out_size, void* d_ws, size_t ws_size,
                              hipStream_t stream) {
  const float*         x       = (const float*)d_in[0];
  const unsigned char* adj     = (const unsigned char*)d_in[1];  // jnp bool -> 1 byte
  const float*         W       = (const float*)d_in[2];
  const float*         att_src = (const float*)d_in[3];
  const float*         att_dst = (const float*)d_in[4];
  float*               out     = (float*)d_out;

  // workspace: hT f16 [B][H][HD][L], a_src f32 [B*H*L], a_dst f32 [B*H*L]
  char* ws = (char*)d_ws;
  _Float16* hT   = (_Float16*)ws;                       // 8.39 MB
  float*    asrc = (float*)(ws + (size_t)BB * NH * HD * LL * sizeof(_Float16));
  float*    adst = asrc + (size_t)BB * NH * LL;         // +256 KB each

  gat_proj<<<dim3((16384 / 16) * (COUT / 16) / 8), dim3(256), 0, stream>>>(x, W, hT);
  gat_attvec<<<dim3(BB * NH * LL / 256), dim3(256), 0, stream>>>(hT, att_src, att_dst,
                                                                 asrc, adst);
  gat_attn<<<dim3(BB * NH * (LL / 32)), dim3(256), 0, stream>>>(adj, hT, asrc, adst, out);
}
